// RandNLAGQALayer_89781996355794
// MI455X (gfx1250) — matrix operand order repaired and checked by reading, not compiled, not run on hardware
//
#include <hip/hip_runtime.h>

// ---------------- constants ----------------
#define S_LEN 8192
#define HIDN  2048
#define NH    16
#define NKV   4
#define HD    128
#define SKN   640
#define TKN   2048
#define TOTN  2688   // TKN + SKN

typedef __attribute__((ext_vector_type(16))) __bf16 v16bf;
typedef __attribute__((ext_vector_type(8)))  float  v8f;

union BF16Frag { v16bf v; unsigned short u[16]; };
union LdU4     { uint4 q[2]; unsigned short u[16]; };

__device__ __forceinline__ unsigned short tobf(float f) {
  unsigned int b = __float_as_uint(f);
  return (unsigned short)((b + 0x7FFFu + ((b >> 16) & 1u)) >> 16);
}

__device__ __forceinline__ v8f wmma_bf16(v16bf a, v16bf b, v8f c) {
  return __builtin_amdgcn_wmma_f32_16x16x32_bf16(
      /*neg_a=*/false, a, /*neg_b=*/false, b,
      /*c_mod=*/(short)0, c, /*reuse_a=*/false, /*reuse_b=*/false);
}

// A-matrix 16x32 bf16 layout (ISA 7.12.2): lane m = lane&15, half = lane>>4
__device__ __forceinline__ int a_kmap(int e, int half) {
  return ((e >> 1) & 3) * 2 + (e & 1) + half * 8 + (e >> 3) * 16;
}

// ---------------- Tensor Data Mover (CDNA5 TDM) ----------------
#if defined(__AMDGCN__) && __has_builtin(__builtin_amdgcn_tensor_load_to_lds) && \
    __has_builtin(__builtin_amdgcn_s_wait_tensorcnt)
#define USE_TDM 1
#else
#define USE_TDM 0
#endif

#if USE_TDM
typedef unsigned int u32x4 __attribute__((ext_vector_type(4)));
typedef int i32x4 __attribute__((ext_vector_type(4)));
typedef int i32x8 __attribute__((ext_vector_type(8)));

// Generic pointer to a __shared__ object: low 32 bits are the LDS byte address.
__device__ __forceinline__ unsigned lds_off(const void* p) {
  return (unsigned)(unsigned long long)p;
}

// 2D tile load, bf16 elements (data_size=2B). Destination LDS gets
// (pad_amount_code+1)*... padding: pad_interval 2^(c+1) DWORDs, pad (c+1) DWORDs.
__device__ __forceinline__ void tdm_load_2d(unsigned lds_addr, const void* gptr,
                                            int tile_w, int tile_h,
                                            int row_stride_elems,
                                            int pad_interval_code,
                                            int pad_amount_code, int pad_en) {
  unsigned long long ga = (unsigned long long)gptr;
  u32x4 g0;
  g0[0] = 1u;                                        // count=1, user descriptor
  g0[1] = lds_addr;                                  // lds_addr [63:32]
  g0[2] = (unsigned)(ga & 0xFFFFFFFFu);              // global_addr lo
  g0[3] = (unsigned)((ga >> 32) & 0x01FFFFFFu) | (2u << 30);  // addr hi | type=2
  i32x8 g1;
  g1[0] = (1 << 16)                                  // data_size = 2 bytes
        | (pad_en << 20)
        | (pad_interval_code << 22)
        | (pad_amount_code << 25);
  g1[1] = (tile_w & 0xFFFF) << 16;                   // tensor_dim0 lo16 @ [63:48]
  g1[2] = ((tile_w >> 16) & 0xFFFF) | ((tile_h & 0xFFFF) << 16);  // d0 hi | d1 lo
  g1[3] = ((tile_h >> 16) & 0xFFFF) | ((tile_w & 0xFFFF) << 16);  // d1 hi | tile_dim0
  g1[4] = (tile_h & 0xFFFF);                         // tile_dim1 (tile_dim2=0)
  g1[5] = row_stride_elems;                          // tensor_dim0_stride lo32
  g1[6] = 0;
  g1[7] = 0;
  i32x4 zz = {0, 0, 0, 0};
#if __clang_major__ >= 23
  i32x8 z8 = {0, 0, 0, 0, 0, 0, 0, 0};
  __builtin_amdgcn_tensor_load_to_lds(g0, g1, zz, zz, z8, 0);
#else
  __builtin_amdgcn_tensor_load_to_lds(g0, g1, zz, zz, 0);
#endif
}
#endif  // USE_TDM

// ---------------- elementwise f32 -> bf16 ----------------
__global__ void f2bf_kernel(const float* __restrict__ in,
                            unsigned short* __restrict__ out, int n) {
  int i = blockIdx.x * 256 + threadIdx.x;
  if (i < n) out[i] = tobf(in[i]);
}

// ---------------- importance MLP ----------------
__global__ void imp_mlp_kernel(const float* __restrict__ x,
                               const float* __restrict__ w1, const float* __restrict__ b1,
                               const float* __restrict__ w2, const float* __restrict__ b2,
                               float* __restrict__ logits, float* __restrict__ wts) {
  __shared__ float xs[HIDN];
  __shared__ float hred[64];
  int t = blockIdx.x;
  for (int i = threadIdx.x; i < HIDN; i += 64) xs[i] = x[(size_t)t * HIDN + i];
  __syncthreads();
  float acc = b1[threadIdx.x];
  for (int i = 0; i < HIDN; ++i) acc += xs[i] * w1[i * 64 + threadIdx.x];
  hred[threadIdx.x] = tanhf(acc) * w2[threadIdx.x];
  __syncthreads();
  if (threadIdx.x == 0) {
    float s = b2[0];
    for (int i = 0; i < 64; ++i) s += hred[i];
    s -= 2.5494451709255714f;  // log(8192/640)
    logits[t] = s;
    wts[t] = 1.f / (1.f + expf(-s));
  }
}

// ---------------- top-k via in-LDS bitonic sort (descending) ----------------
__global__ __launch_bounds__(1024) void topk_kernel(
    const float* __restrict__ logits, const float* __restrict__ wts,
    int* __restrict__ idx_out, int* __restrict__ mask, float* __restrict__ restw) {
  __shared__ float v[S_LEN];
  __shared__ int   id[S_LEN];
  int tid = threadIdx.x;
  for (int i = tid; i < S_LEN; i += 1024) { v[i] = logits[i]; id[i] = i; }
  __syncthreads();
  for (int size = 2; size <= S_LEN; size <<= 1) {
    for (int stride = size >> 1; stride > 0; stride >>= 1) {
      for (int p = tid; p < S_LEN / 2; p += 1024) {
        int i = (p / stride) * (stride * 2) + (p % stride);
        int j = i + stride;
        bool desc = ((i & size) == 0);
        float vi = v[i], vj = v[j];
        int   ii = id[i], jj = id[j];
        bool sw = desc ? (vi < vj || (vi == vj && ii > jj))
                       : (vi > vj || (vi == vj && ii < jj));
        if (sw) { v[i] = vj; v[j] = vi; id[i] = jj; id[j] = ii; }
      }
      __syncthreads();
    }
  }
  for (int i = tid; i < TKN; i += 1024) idx_out[i] = id[i];
  for (int i = tid; i < S_LEN; i += 1024) mask[i] = 0;
  __syncthreads();
  for (int i = tid; i < TKN; i += 1024) mask[id[i]] = 1;
  __syncthreads();
  for (int i = tid; i < S_LEN; i += 1024)
    restw[i] = wts[i] * (mask[i] ? 0.f : 1.f);
}

// ---------------- bf16 WMMA GEMM: C(MxN) = A(MxK) @ B(KxN), row-major ----------------
// TDM double-buffered: wave0 issues next tile's tensor_load_to_lds while WMMA
// runs on the current buffer; s_wait_tensorcnt(2) keeps the prefetch in flight.
__global__ __launch_bounds__(256) void gemm_bf16_kernel(
    const unsigned short* __restrict__ A, const unsigned short* __restrict__ B,
    float* __restrict__ C, int M, int N, int K) {
  __shared__ unsigned short As[2][128][36];   // 32-half rows + 4-half pad
  __shared__ unsigned short Bs[2][32][132];   // 128-half rows + 4-half pad
  const int tid = threadIdx.x;
  const int wave = tid >> 5, lane = tid & 31;
  const int half = lane >> 4, l16 = lane & 15;
  const int wm = wave >> 1, wn = wave & 1;
  const int bm = blockIdx.y * 128, bn = blockIdx.x * 128;
  v8f acc[2][4];
  v8f z = {};
  for (int i = 0; i < 2; ++i) for (int j = 0; j < 4; ++j) acc[i][j] = z;

  const int nk = K / 32;
#if USE_TDM
  if (tid == 0) {
    tdm_load_2d(lds_off(&As[0][0][0]), A + (size_t)bm * K, 32, 128, K, 3, 1, 1);
    tdm_load_2d(lds_off(&Bs[0][0][0]), B + bn, 128, 32, N, 5, 1, 1);
  }
#endif
  for (int ks = 0; ks < nk; ++ks) {
    const int cur = ks & 1;
#if USE_TDM
    if (tid == 0) {
      if (ks + 1 < nk) {
        tdm_load_2d(lds_off(&As[cur ^ 1][0][0]),
                    A + (size_t)bm * K + (ks + 1) * 32, 32, 128, K, 3, 1, 1);
        tdm_load_2d(lds_off(&Bs[cur ^ 1][0][0]),
                    B + (size_t)(ks + 1) * 32 * N + bn, 128, 32, N, 5, 1, 1);
        __builtin_amdgcn_s_wait_tensorcnt(2);   // current tile's 2 DMAs done
      } else {
        __builtin_amdgcn_s_wait_tensorcnt(0);
      }
    }
    __syncthreads();
#else
    {
      int r = tid >> 1, c = (tid & 1) * 16;
      LdU4 t;
      const uint4* src = (const uint4*)(A + (size_t)(bm + r) * K + ks * 32 + c);
      t.q[0] = src[0]; t.q[1] = src[1];
#pragma unroll
      for (int e = 0; e < 16; ++e) As[cur][r][c + e] = t.u[e];
    }
    {
      int r = tid >> 3, c = (tid & 7) * 16;
      LdU4 t;
      const uint4* src = (const uint4*)(B + (size_t)(ks * 32 + r) * N + bn + c);
      t.q[0] = src[0]; t.q[1] = src[1];
#pragma unroll
      for (int e = 0; e < 16; ++e) Bs[cur][r][c + e] = t.u[e];
    }
    __syncthreads();
#endif
    BF16Frag afr[2], bfr[4];
#pragma unroll
    for (int mt = 0; mt < 2; ++mt) {
      int m = wm * 32 + mt * 16 + l16;
#pragma unroll
      for (int e = 0; e < 16; ++e) afr[mt].u[e] = As[cur][m][a_kmap(e, half)];
    }
#pragma unroll
    for (int nt = 0; nt < 4; ++nt) {
      int n0 = wn * 64 + nt * 16;
#pragma unroll
      for (int e = 0; e < 16; ++e) bfr[nt].u[e] = Bs[cur][a_kmap(e, half)][n0 + l16];
    }
#pragma unroll
    for (int mt = 0; mt < 2; ++mt)
#pragma unroll
      for (int nt = 0; nt < 4; ++nt)
        acc[mt][nt] = wmma_bf16(afr[mt].v, bfr[nt].v, acc[mt][nt]);
    __syncthreads();
  }
#pragma unroll
  for (int mt = 0; mt < 2; ++mt)
#pragma unroll
    for (int nt = 0; nt < 4; ++nt)
#pragma unroll
      for (int r = 0; r < 8; ++r) {
        int m = bm + wm * 32 + mt * 16 + r + half * 8;
        int n = bn + wn * 64 + nt * 16 + l16;
        C[(size_t)m * N + n] = acc[mt][nt][r];
      }
}

// ---------------- RMS norm over last dim (128), one wave per row ----------------
__global__ void rms_kernel(float* __restrict__ x, const float* __restrict__ w, int rows) {
  int row = blockIdx.x * 8 + (threadIdx.x >> 5);
  int lane = threadIdx.x & 31;
  if (row >= rows) return;
  float* p = x + (size_t)row * HD;
  float v4[4], s = 0.f;
#pragma unroll
  for (int i = 0; i < 4; ++i) { v4[i] = p[lane + i * 32]; s += v4[i] * v4[i]; }
#pragma unroll
  for (int m = 16; m >= 1; m >>= 1) s += __shfl_xor(s, m, 32);
  float inv = rsqrtf(s * (1.f / 128.f) + 1e-6f);
#pragma unroll
  for (int i = 0; i < 4; ++i) p[lane + i * 32] = v4[i] * inv * w[lane + i * 32];
}

// ---------------- gather top-k rows, apply RoPE, emit bf16 q_f/k_f/v_f ----------------
__global__ void build_det_kernel(const float* __restrict__ q, const float* __restrict__ k,
                                 const float* __restrict__ v,
                                 const float* __restrict__ cosb, const float* __restrict__ sinb,
                                 const int* __restrict__ idx,
                                 unsigned short* __restrict__ qf,
                                 unsigned short* __restrict__ kf,
                                 unsigned short* __restrict__ vf) {
  int j = blockIdx.x;
  int t = idx[j];
  for (int c = threadIdx.x; c < NH * HD; c += 256) {
    int d = c & (HD - 1);
    float cv = cosb[(size_t)t * HD + d], sv = sinb[(size_t)t * HD + d];
    float a = q[(size_t)t * (NH * HD) + c];
    float b = (d < 64) ? -q[(size_t)t * (NH * HD) + c + 64]
                       :  q[(size_t)t * (NH * HD) + c - 64];
    qf[(size_t)j * (NH * HD) + c] = tobf(a * cv + b * sv);
  }
  for (int c = threadIdx.x; c < NKV * HD; c += 256) {
    int d = c & (HD - 1);
    float cv = cosb[(size_t)t * HD + d], sv = sinb[(size_t)t * HD + d];
    float a = k[(size_t)t * (NKV * HD) + c];
    float b = (d < 64) ? -k[(size_t)t * (NKV * HD) + c + 64]
                       :  k[(size_t)t * (NKV * HD) + c - 64];
    kf[(size_t)j * (NKV * HD) + c] = tobf(a * cv + b * sv);
    vf[(size_t)j * (NKV * HD) + c] = tobf(v[(size_t)t * (NKV * HD) + c]);
  }
}

// ---------------- Kronecker DCT sketch: one block per (head,d) signal ----------------
__global__ void sketch_kernel(const float* __restrict__ x, const float* __restrict__ w,
                              const float* __restrict__ ka, const float* __restrict__ kb,
                              const float* __restrict__ sscale, int nh,
                              unsigned short* __restrict__ out) {
  __shared__ float xs[S_LEN];
  __shared__ float tmp[32][33];
  int hd = blockIdx.x;
  int stride = nh * HD;
  for (int t = threadIdx.x; t < S_LEN; t += 256)
    xs[t] = x[(size_t)t * stride + hd] * w[t];
  __syncthreads();
  for (int p = threadIdx.x; p < 1024; p += 256) {
    int i = p >> 5, kk = p & 31;
    const float* xp = xs + i * 256;
    const float* kp = kb + kk * 256;
    float s = 0.f;
    for (int j = 0; j < 256; ++j) s += xp[j] * kp[j];
    tmp[i][kk] = s;
  }
  __syncthreads();
  float sc = sscale[0];
  for (int c = threadIdx.x; c < SKN; c += 256) {
    int kk = c / 20, a = c % 20;
    float s = 0.f;
    for (int i = 0; i < 32; ++i) s += tmp[i][kk] * ka[a * HD + i];
    out[(size_t)(TKN + c) * stride + hd] = tobf(s * sc);
  }
}

// ---------------- flash attention: 128 q-rows per block, online softmax ----------------
__global__ __launch_bounds__(256) void flash_attn_kernel(
    const unsigned short* __restrict__ qf, const unsigned short* __restrict__ kf,
    const unsigned short* __restrict__ vf, float* __restrict__ attn) {
  __shared__ unsigned short Ks[128][132];
  __shared__ unsigned short Vs[128][132];
  __shared__ unsigned short Ps[8][16][132];
  const int h = blockIdx.y;
  const int q0 = blockIdx.x * 128;
  const int kvh = h >> 2;                       // GQA repeat: H/KV = 4
  const int tid = threadIdx.x, wave = tid >> 5, lane = tid & 31;
  const int half = lane >> 4, l16 = lane & 15;

  BF16Frag qfr[4];
  {
    int m = q0 + wave * 16 + l16;
    const unsigned short* qp = qf + (size_t)m * (NH * HD) + h * HD;
#pragma unroll
    for (int dc = 0; dc < 4; ++dc)
#pragma unroll
      for (int e = 0; e < 16; ++e) qfr[dc].u[e] = qp[dc * 32 + a_kmap(e, half)];
  }

  v8f o[8];
  v8f z = {};
  for (int i = 0; i < 8; ++i) o[i] = z;
  float mrow[8], lrow[8];
#pragma unroll
  for (int r = 0; r < 8; ++r) { mrow[r] = -1e30f; lrow[r] = 0.f; }
  const float scale = 0.08838834764831845f;     // 1/sqrt(128)

  for (int kt = 0; kt < TOTN / 128; ++kt) {
#if USE_TDM
    if (tid == 0) {
      tdm_load_2d(lds_off(&Ks[0][0]),
                  kf + (size_t)(kt * 128) * (NKV * HD) + kvh * HD,
                  128, 128, NKV * HD, 5, 1, 1);
      tdm_load_2d(lds_off(&Vs[0][0]),
                  vf + (size_t)(kt * 128) * (NKV * HD) + kvh * HD,
                  128, 128, NKV * HD, 5, 1, 1);
      __builtin_amdgcn_s_wait_tensorcnt(0);
    }
    __syncthreads();
#else
    for (int p = tid; p < 128 * 8; p += 256) {
      int r = p >> 3, c = (p & 7) * 16;
      LdU4 tk, tv;
      const uint4* ks = (const uint4*)(kf + (size_t)(kt * 128 + r) * (NKV * HD) + kvh * HD + c);
      const uint4* vs = (const uint4*)(vf + (size_t)(kt * 128 + r) * (NKV * HD) + kvh * HD + c);
      tk.q[0] = ks[0]; tk.q[1] = ks[1];
      tv.q[0] = vs[0]; tv.q[1] = vs[1];
#pragma unroll
      for (int e = 0; e < 16; ++e) { Ks[r][c + e] = tk.u[e]; Vs[r][c + e] = tv.u[e]; }
    }
    __syncthreads();
#endif

    v8f sacc[8];
    for (int i = 0; i < 8; ++i) sacc[i] = z;
#pragma unroll
    for (int nt = 0; nt < 8; ++nt) {
#pragma unroll
      for (int dc = 0; dc < 4; ++dc) {
        BF16Frag bfr;  // B[dd][n] = K[nt*16+n][dc*32+dd]
#pragma unroll
        for (int e = 0; e < 16; ++e) bfr.u[e] = Ks[nt * 16 + l16][dc * 32 + a_kmap(e, half)];
        sacc[nt] = wmma_bf16(qfr[dc].v, bfr.v, sacc[nt]);
      }
    }
#pragma unroll
    for (int nt = 0; nt < 8; ++nt)
#pragma unroll
      for (int r = 0; r < 8; ++r) sacc[nt][r] *= scale;
#pragma unroll
    for (int r = 0; r < 8; ++r) {
      float mx = sacc[0][r];
#pragma unroll
      for (int nt = 1; nt < 8; ++nt) mx = fmaxf(mx, sacc[nt][r]);
#pragma unroll
      for (int msk = 8; msk >= 1; msk >>= 1) mx = fmaxf(mx, __shfl_xor(mx, msk, 16));
      float mnew = fmaxf(mrow[r], mx);
      float alpha = __expf(mrow[r] - mnew);
      float rs = 0.f;
#pragma unroll
      for (int nt = 0; nt < 8; ++nt) {
        float pv = __expf(sacc[nt][r] - mnew);
        sacc[nt][r] = pv;
        rs += pv;
      }
#pragma unroll
      for (int msk = 8; msk >= 1; msk >>= 1) rs += __shfl_xor(rs, msk, 16);
      lrow[r] = lrow[r] * alpha + rs;
      mrow[r] = mnew;
#pragma unroll
      for (int dt = 0; dt < 8; ++dt) o[dt][r] *= alpha;
    }
#pragma unroll
    for (int nt = 0; nt < 8; ++nt)
#pragma unroll
      for (int r = 0; r < 8; ++r)
        Ps[wave][r + half * 8][nt * 16 + l16] = tobf(sacc[nt][r]);
#pragma unroll
    for (int dt = 0; dt < 8; ++dt) {
#pragma unroll
      for (int kc = 0; kc < 4; ++kc) {
        BF16Frag afr, bfr;
#pragma unroll
        for (int e = 0; e < 16; ++e) afr.u[e] = Ps[wave][l16][kc * 32 + a_kmap(e, half)];
#pragma unroll
        for (int e = 0; e < 16; ++e) bfr.u[e] = Vs[kc * 32 + a_kmap(e, half)][dt * 16 + l16];
        o[dt] = wmma_bf16(afr.v, bfr.v, o[dt]);
      }
    }
    __syncthreads();
  }
#pragma unroll
  for (int dt = 0; dt < 8; ++dt)
#pragma unroll
    for (int r = 0; r < 8; ++r) {
      int m = q0 + wave * 16 + r + half * 8;
      attn[(size_t)m * (NH * HD) + h * HD + dt * 16 + l16] = o[dt][r] / lrow[r];
    }
}

// ---------------- Kronecker expand: 640 coeffs -> 8192 sequence, per (h,d) ----------------
__global__ void expand_kernel(const float* __restrict__ attn,
                              const float* __restrict__ ka, const float* __restrict__ kb,
                              float* __restrict__ full) {
  __shared__ float coef[32][21];
  __shared__ float g[32][33];
  int hd = blockIdx.x;
  for (int c = threadIdx.x; c < SKN; c += 256)
    coef[c / 20][c % 20] = attn[(size_t)(TKN + c) * (NH * HD) + hd];
  __syncthreads();
  for (int p = threadIdx.x; p < 1024; p += 256) {
    int kk = p >> 5, i = p & 31;
    float s = 0.f;
    for (int a = 0; a < 20; ++a) s += coef[kk][a] * ka[a * HD + i];
    g[kk][i] = s;
  }
  __syncthreads();
  for (int t = threadIdx.x; t < S_LEN; t += 256) {
    int i = t >> 8, jj = t & 255;
    float s = 0.f;
    for (int kk = 0; kk < 32; ++kk) s += g[kk][i] * kb[kk * 256 + jj];
    full[(size_t)t * (NH * HD) + hd] = s;
  }
}

// ---------------- overwrite top-k rows with deterministic attention output ----------------
__global__ void fixup_kernel(const float* __restrict__ attn, const int* __restrict__ idx,
                             float* __restrict__ full) {
  int j = blockIdx.x;
  int t = idx[j];
  for (int c = threadIdx.x; c < NH * HD; c += 256)
    full[(size_t)t * (NH * HD) + c] = attn[(size_t)j * (NH * HD) + c];
}

// ---------------- host launcher ----------------
extern "C" void kernel_launch(void* const* d_in, const int* in_sizes, int n_in,
                              void* d_out, int out_size, void* d_ws, size_t ws_size,
                              hipStream_t stream) {
  const float* x     = (const float*)d_in[0];
  const float* cosb  = (const float*)d_in[1];
  const float* sinb  = (const float*)d_in[2];
  const float* wq    = (const float*)d_in[3];
  const float* wk    = (const float*)d_in[4];
  const float* wv    = (const float*)d_in[5];
  const float* wo    = (const float*)d_in[6];
  const float* qnw   = (const float*)d_in[7];
  const float* knw   = (const float*)d_in[8];
  const float* iw1   = (const float*)d_in[9];
  const float* ib1   = (const float*)d_in[10];
  const float* iw2   = (const float*)d_in[11];
  const float* ib2   = (const float*)d_in[12];
  const float* ka    = (const float*)d_in[13];
  const float* kb    = (const float*)d_in[14];
  const float* sscale= (const float*)d_in[15];

  char* ws = (char*)d_ws;
  size_t off = 0;
  auto alloc = [&](size_t bytes) {
    void* p = ws + off;
    off = (off + bytes + 255) & ~(size_t)255;
    return p;
  };
  unsigned short* xbf  = (unsigned short*)alloc((size_t)S_LEN * HIDN * 2); // reused as full_bf
  unsigned short* wqbf = (unsigned short*)alloc((size_t)HIDN * (NH * HD) * 2);
  unsigned short* wkbf = (unsigned short*)alloc((size_t)HIDN * (NKV * HD) * 2);
  unsigned short* wvbf = (unsigned short*)alloc((size_t)HIDN * (NKV * HD) * 2);
  unsigned short* wobf = (unsigned short*)alloc((size_t)(NH * HD) * HIDN * 2);
  float* logits = (float*)alloc(S_LEN * 4);
  float* wts    = (float*)alloc(S_LEN * 4);
  float* restw  = (float*)alloc(S_LEN * 4);
  int*   idx    = (int*)alloc(TKN * 4);
  int*   mask   = (int*)alloc(S_LEN * 4);
  float* qfull  = (float*)alloc((size_t)S_LEN * (NH * HD) * 4);
  float* kfull  = (float*)alloc((size_t)S_LEN * (NKV * HD) * 4);
  float* vfull  = (float*)alloc((size_t)S_LEN * (NKV * HD) * 4);
  unsigned short* qfb = (unsigned short*)alloc((size_t)TOTN * (NH * HD) * 2);
  unsigned short* kfb = (unsigned short*)alloc((size_t)TOTN * (NKV * HD) * 2);
  unsigned short* vfb = (unsigned short*)alloc((size_t)TOTN * (NKV * HD) * 2);
  float* attn  = (float*)alloc((size_t)TOTN * (NH * HD) * 4);
  float* full  = (float*)alloc((size_t)S_LEN * (NH * HD) * 4);
  (void)ws_size; (void)in_sizes; (void)n_in; (void)out_size;

  auto cvt = [&](const float* src, unsigned short* dst, int n) {
    f2bf_kernel<<<(n + 255) / 256, 256, 0, stream>>>(src, dst, n);
  };
  cvt(x,  xbf,  S_LEN * HIDN);
  cvt(wq, wqbf, HIDN * NH * HD);
  cvt(wk, wkbf, HIDN * NKV * HD);
  cvt(wv, wvbf, HIDN * NKV * HD);
  cvt(wo, wobf, NH * HD * HIDN);

  imp_mlp_kernel<<<S_LEN, 64, 0, stream>>>(x, iw1, ib1, iw2, ib2, logits, wts);
  topk_kernel<<<1, 1024, 0, stream>>>(logits, wts, idx, mask, restw);

  gemm_bf16_kernel<<<dim3(16, 64), 256, 0, stream>>>(xbf, wqbf, qfull, S_LEN, NH * HD, HIDN);
  gemm_bf16_kernel<<<dim3(4, 64), 256, 0, stream>>>(xbf, wkbf, kfull, S_LEN, NKV * HD, HIDN);
  gemm_bf16_kernel<<<dim3(4, 64), 256, 0, stream>>>(xbf, wvbf, vfull, S_LEN, NKV * HD, HIDN);

  rms_kernel<<<(S_LEN * NH + 7) / 8, 256, 0, stream>>>(qfull, qnw, S_LEN * NH);
  rms_kernel<<<(S_LEN * NKV + 7) / 8, 256, 0, stream>>>(kfull, knw, S_LEN * NKV);

  build_det_kernel<<<TKN, 256, 0, stream>>>(qfull, kfull, vfull, cosb, sinb, idx, qfb, kfb, vfb);
  sketch_kernel<<<NH * HD, 256, 0, stream>>>(qfull, wts,   ka, kb, sscale, NH,  qfb);
  sketch_kernel<<<NKV * HD, 256, 0, stream>>>(kfull, restw, ka, kb, sscale, NKV, kfb);
  sketch_kernel<<<NKV * HD, 256, 0, stream>>>(vfull, restw, ka, kb, sscale, NKV, vfb);

  flash_attn_kernel<<<dim3(TOTN / 128, NH), 256, 0, stream>>>(qfb, kfb, vfb, attn);

  expand_kernel<<<NH * HD, 256, 0, stream>>>(attn, ka, kb, full);
  fixup_kernel<<<TKN, 256, 0, stream>>>(attn, idx, full);

  cvt(full, xbf, S_LEN * NH * HD);
  gemm_bf16_kernel<<<dim3(16, 64), 256, 0, stream>>>(xbf, wobf, (float*)d_out,
                                                     S_LEN, HIDN, NH * HD);
}
